// MaskFormerSwinLayer_4337916969424
// MI455X (gfx1250) — matrix operand order repaired and checked by reading, not compiled, not run on hardware
//
#include <hip/hip_runtime.h>
#include <math.h>

// ---------------------------------------------------------------------------
// Swin block for gfx1250: all GEMMs on v_wmma_f32_16x16x32_bf16 (wave32).
// GEMM is register-blocked: 64x64 output per wave -> 1 b128 load per WMMA.
// ---------------------------------------------------------------------------

typedef __attribute__((ext_vector_type(16))) __bf16 v16bf;
typedef __attribute__((ext_vector_type(8)))  float  v8f;

union FragBF { uint4 u4[2]; v16bf v; unsigned short s[16]; };

__device__ __forceinline__ unsigned short f2bf(float f) {
  unsigned int u = __float_as_uint(f);
  u += 0x7FFFu + ((u >> 16) & 1u);
  return (unsigned short)(u >> 16);
}

#define DIM      384
#define NH       12
#define HD       32
#define WIN      7
#define NTOK     49           // 7*7
#define NPAD     64           // padded window rows
#define SHIFT    3
#define IMG      56
#define BATCH    8
#define LSEQ     (IMG*IMG)    // 3136
#define NWIN     (BATCH*64)   // 512 windows
#define MWROWS   (NWIN*NPAD)  // 32768 padded window rows
#define MROWS    (BATCH*LSEQ) // 25088 real rows
#define FFN      (4*DIM)      // 1536

// ---------------------------------------------------------------------------
// fp32 [K][N] weight -> bf16 transposed [N][K]  (B-fragment wants column-major)
// ---------------------------------------------------------------------------
__global__ void cvt_transpose_bf16(const float* __restrict__ src,
                                   unsigned short* __restrict__ dst,
                                   int K, int N) {
  int i = blockIdx.x * blockDim.x + threadIdx.x;
  if (i >= K * N) return;
  int k = i / N, n = i - k * N;
  dst[(size_t)n * K + k] = f2bf(src[i]);
}

// ---------------------------------------------------------------------------
// LN1 + roll(-3,-3) + 7x7 window partition -> xw bf16 [NWIN][64][384]
// one block (128 thr) per padded window row
// ---------------------------------------------------------------------------
__global__ __launch_bounds__(128)
void ln1_window_kernel(const float* __restrict__ hs,
                       const float* __restrict__ g,
                       const float* __restrict__ b,
                       unsigned short* __restrict__ xw) {
  int blk = blockIdx.x;            // win*64 + row
  int win = blk >> 6, row = blk & 63;
  int tid = threadIdx.x;
  unsigned short* dst = xw + (size_t)blk * DIM;
  if (row >= NTOK) {               // zero pad rows (whole block uniform)
    for (int c = tid; c < DIM; c += 128) dst[c] = 0;
    return;
  }
  int bI = win >> 6, wi = win & 63, wy = wi >> 3, wx = wi & 7;
  int iy = row / WIN, ix = row - iy * WIN;
  int sy = (wy * WIN + iy + SHIFT) % IMG;   // roll(-s): rolled[i]=orig[(i+s)%56]
  int sx = (wx * WIN + ix + SHIFT) % IMG;
  const float* src = hs + ((size_t)bI * LSEQ + sy * IMG + sx) * DIM;
  float v0 = src[tid], v1 = src[tid + 128], v2 = src[tid + 256];
  __shared__ float rs[128], rq[128];
  rs[tid] = v0 + v1 + v2;
  rq[tid] = v0 * v0 + v1 * v1 + v2 * v2;
  __syncthreads();
  for (int o = 64; o > 0; o >>= 1) {
    if (tid < o) { rs[tid] += rs[tid + o]; rq[tid] += rq[tid + o]; }
    __syncthreads();
  }
  float mu  = rs[0] * (1.0f / DIM);
  float var = rq[0] * (1.0f / DIM) - mu * mu;
  float inv = rsqrtf(var + 1e-5f);
  dst[tid]       = f2bf((v0 - mu) * inv * g[tid]       + b[tid]);
  dst[tid + 128] = f2bf((v1 - mu) * inv * g[tid + 128] + b[tid + 128]);
  dst[tid + 256] = f2bf((v2 - mu) * inv * g[tid + 256] + b[tid + 256]);
}

// ---------------------------------------------------------------------------
// Register-blocked bf16 WMMA GEMM: out[M][N] = A[M][K] @ Bt[N][K]^T + bias.
// One wave computes a 64x64 macro-tile: 4x4 v8f accumulators (128 VGPRs).
// Per 32-deep K-step: 4 A frags + 4 B frags (16 b128 loads) -> 16 WMMAs.
// ---------------------------------------------------------------------------
#define GEMM_OUT_BF16      0
#define GEMM_OUT_BF16_GELU 1
#define GEMM_OUT_F32       2
#define GEMM_OUT_F32_RES   3

__global__ __launch_bounds__(256)
void gemm_bf16_wmma(const unsigned short* __restrict__ A,
                    const unsigned short* __restrict__ Bt,
                    const float* __restrict__ bias,
                    void* __restrict__ out,
                    const float* __restrict__ resid,
                    int M, int N, int K, int mode) {
  int tileId = blockIdx.x * (blockDim.x >> 5) + (threadIdx.x >> 5);
  int nT = N >> 6;
  if (tileId >= (M >> 6) * nT) return;       // wave-uniform
  int mt = tileId / nT, nt = tileId - mt * nT;
  int lane = threadIdx.x & 31, sel = lane >> 4, lm = lane & 15;

  const unsigned short* aBase = A  + (size_t)(mt * 64 + lm) * K;
  const unsigned short* bBase = Bt + (size_t)(nt * 64 + lm) * K;

  v8f acc[4][4] = {};
  for (int kb = 0; kb < K; kb += 32) {
    FragBF fa[4];
#pragma unroll
    for (int i = 0; i < 4; ++i) {
      // A 16x32 bf16 layout: lane holds row lm; K chunks split by sel
      fa[i].u4[0] = *(const uint4*)(aBase + (size_t)i * 16 * K + kb + sel * 8);
      fa[i].u4[1] = *(const uint4*)(aBase + (size_t)i * 16 * K + kb + 16 + sel * 8);
    }
#pragma unroll
    for (int j = 0; j < 4; ++j) {
      FragBF fb;  // B 32x16 (column-major source): K chunks split by sel, same as A
      fb.u4[0] = *(const uint4*)(bBase + (size_t)j * 16 * K + kb + sel * 8);
      fb.u4[1] = *(const uint4*)(bBase + (size_t)j * 16 * K + kb + 16 + sel * 8);
#pragma unroll
      for (int i = 0; i < 4; ++i)
        acc[i][j] = __builtin_amdgcn_wmma_f32_16x16x32_bf16(
            false, fa[i].v, false, fb.v, (short)0, acc[i][j], false, false);
    }
  }

#pragma unroll
  for (int j = 0; j < 4; ++j) {
    int col = nt * 64 + j * 16 + lm;
    float bv = bias ? bias[col] : 0.0f;
#pragma unroll
    for (int i = 0; i < 4; ++i) {
#pragma unroll
      for (int r = 0; r < 8; ++r) {
        int orow = mt * 64 + i * 16 + sel * 8 + r;   // C/D: M = r + 8*sel
        size_t idx = (size_t)orow * N + col;
        float x = acc[i][j][r] + bv;
        if (mode == GEMM_OUT_BF16_GELU)
          x = 0.5f * x * (1.0f + erff(x * 0.70710678118654752f));
        if (mode <= GEMM_OUT_BF16_GELU)
          ((unsigned short*)out)[idx] = f2bf(x);
        else if (mode == GEMM_OUT_F32_RES)
          ((float*)out)[idx] = x + resid[idx];
        else
          ((float*)out)[idx] = x;
      }
    }
  }
}

// ---------------------------------------------------------------------------
// Window attention: one block (4 waves) per (window, head).
// scores: 16 WMMA tiles, K=32=hd (one v_wmma each); softmax fp32 in LDS;
// ctx = probs@V: 8 tiles x 2 K-steps; probs A-frags via ds_load_b128.
// ---------------------------------------------------------------------------
__global__ __launch_bounds__(128)
void attn_kernel(const unsigned short* __restrict__ Q,
                 const unsigned short* __restrict__ Kk,
                 const unsigned short* __restrict__ V,
                 const float* __restrict__ rpb,
                 const int* __restrict__ rpi,
                 unsigned short* __restrict__ CTX) {
  int win = blockIdx.x, head = blockIdx.y;
  __shared__ float          sc[NPAD][NPAD];
  __shared__ unsigned short pr[NPAD][NPAD];
  int tid = threadIdx.x, wave = tid >> 5, lane = tid & 31;
  int sel = lane >> 4, lm = lane & 15;
  size_t base = (size_t)win * NPAD * DIM + head * HD;  // row-major [row][head*32+d]

  // ---- scores = Q @ K^T * scale -------------------------------------------
  const int mt = wave;
  for (int nt = 0; nt < 4; ++nt) {
    FragBF fa, fb;
    const unsigned short* qp = Q  + base + (size_t)(mt * 16 + lm) * DIM;
    fa.u4[0] = *(const uint4*)(qp + sel * 8);
    fa.u4[1] = *(const uint4*)(qp + 16 + sel * 8);
    const unsigned short* kp = Kk + base + (size_t)(nt * 16 + lm) * DIM;
    fb.u4[0] = *(const uint4*)(kp + sel * 8);
    fb.u4[1] = *(const uint4*)(kp + 16 + sel * 8);
    v8f acc = {};
    acc = __builtin_amdgcn_wmma_f32_16x16x32_bf16(false, fa.v, false, fb.v,
                                                  (short)0, acc, false, false);
#pragma unroll
    for (int r = 0; r < 8; ++r)
      sc[mt * 16 + sel * 8 + r][nt * 16 + lm] = acc[r] * 0.17677669529663689f;
  }
  __syncthreads();

  // ---- +bias +shift-mask, softmax over valid 49 keys ----------------------
  if (tid < NPAD) {
    int i = tid;
    if (i < NTOK) {
      int wi = win & 63, wy = wi >> 3, wx = wi & 7;
      int iy = i / WIN, ix = i - iy * WIN;
      int gy = wy * WIN + iy, gx = wx * WIN + ix;
      int ri = (gy < 49 ? 0 : (gy < 53 ? 1 : 2)) * 3 +
               (gx < 49 ? 0 : (gx < 53 ? 1 : 2));
      float mx = -1e30f;
      for (int j = 0; j < NTOK; ++j) {
        int idx = rpi[i * NTOK + j];
        float bvs = rpb[idx * NH + head];
        int jy = j / WIN, jx = j - jy * WIN;
        int hy = wy * WIN + jy, hx = wx * WIN + jx;
        int rj = (hy < 49 ? 0 : (hy < 53 ? 1 : 2)) * 3 +
                 (hx < 49 ? 0 : (hx < 53 ? 1 : 2));
        float s = sc[i][j] + bvs + ((ri != rj) ? -100.0f : 0.0f);
        sc[i][j] = s;
        mx = fmaxf(mx, s);
      }
      float sum = 0.0f;
      for (int j = 0; j < NTOK; ++j) {
        float e = __expf(sc[i][j] - mx);
        sc[i][j] = e;
        sum += e;
      }
      float inv = 1.0f / sum;
      for (int j = 0; j < NPAD; ++j)
        pr[i][j] = (j < NTOK) ? f2bf(sc[i][j] * inv) : 0;
    } else {
      for (int j = 0; j < NPAD; ++j) pr[i][j] = 0;  // pad query rows
    }
  }
  __syncthreads();

  // ---- ctx = probs @ V  (64x32, K=64 -> 2 wmma steps per tile) ------------
  for (int t = wave; t < 8; t += 4) {
    int cmt = t >> 1, cnt = t & 1;
    v8f acc = {};
    for (int kb = 0; kb < NPAD; kb += 32) {
      FragBF fa, fb;
      const unsigned short* pp = &pr[cmt * 16 + lm][0];
      fa.u4[0] = *(const uint4*)(pp + kb + sel * 8);        // ds_load_b128
      fa.u4[1] = *(const uint4*)(pp + kb + 16 + sel * 8);
      int d = cnt * 16 + lm;
#pragma unroll
      for (int j = 0; j < 16; ++j)                          // V column gather (A k order)
        fb.s[j] = V[base + (size_t)(kb + sel * 8 + j + (j >= 8 ? 8 : 0)) * DIM + d];
      acc = __builtin_amdgcn_wmma_f32_16x16x32_bf16(false, fa.v, false, fb.v,
                                                    (short)0, acc, false, false);
    }
#pragma unroll
    for (int r = 0; r < 8; ++r)
      CTX[base + (size_t)(cmt * 16 + sel * 8 + r) * DIM + cnt * 16 + lm] =
          f2bf(acc[r]);
  }
}

// ---------------------------------------------------------------------------
// window-reverse + roll(+3,+3) + residual -> h (f32), fused LN2 -> y2 (bf16)
// one block (128 thr) per output token
// ---------------------------------------------------------------------------
__global__ __launch_bounds__(128)
void gather_ln2_kernel(const float* __restrict__ hs,
                       const float* __restrict__ proj,
                       const float* __restrict__ g2,
                       const float* __restrict__ b2,
                       float* __restrict__ h,
                       unsigned short* __restrict__ y2) {
  int blk = blockIdx.x;                 // b*3136 + l
  int bI = blk / LSEQ, l = blk - bI * LSEQ;
  int y = l / IMG, x = l - y * IMG;
  int ry = (y + IMG - SHIFT) % IMG;     // roll(+s): out[i] = rolled[(i-s)%56]
  int rx = (x + IMG - SHIFT) % IMG;
  int wy = ry / WIN, iy = ry - wy * WIN;
  int wx = rx / WIN, ix = rx - wx * WIN;
  size_t srow = ((size_t)(bI * 64 + wy * 8 + wx) * NPAD + iy * WIN + ix) * DIM;
  const float* pv = proj + srow;
  const float* sv = hs + (size_t)blk * DIM;
  int tid = threadIdx.x;
  float h0 = sv[tid]       + pv[tid];
  float h1 = sv[tid + 128] + pv[tid + 128];
  float h2 = sv[tid + 256] + pv[tid + 256];
  float* hd = h + (size_t)blk * DIM;
  hd[tid] = h0; hd[tid + 128] = h1; hd[tid + 256] = h2;
  __shared__ float rs[128], rq[128];
  rs[tid] = h0 + h1 + h2;
  rq[tid] = h0 * h0 + h1 * h1 + h2 * h2;
  __syncthreads();
  for (int o = 64; o > 0; o >>= 1) {
    if (tid < o) { rs[tid] += rs[tid + o]; rq[tid] += rq[tid + o]; }
    __syncthreads();
  }
  float mu  = rs[0] * (1.0f / DIM);
  float var = rq[0] * (1.0f / DIM) - mu * mu;
  float inv = rsqrtf(var + 1e-5f);
  unsigned short* yd = y2 + (size_t)blk * DIM;
  yd[tid]       = f2bf((h0 - mu) * inv * g2[tid]       + b2[tid]);
  yd[tid + 128] = f2bf((h1 - mu) * inv * g2[tid + 128] + b2[tid + 128]);
  yd[tid + 256] = f2bf((h2 - mu) * inv * g2[tid + 256] + b2[tid + 256]);
}

// ---------------------------------------------------------------------------
extern "C" void kernel_launch(void* const* d_in, const int* in_sizes, int n_in,
                              void* d_out, int out_size, void* d_ws, size_t ws_size,
                              hipStream_t stream) {
  (void)in_sizes; (void)n_in; (void)out_size; (void)ws_size;
  const float* hs     = (const float*)d_in[0];
  const float* ln1_g  = (const float*)d_in[1];
  const float* ln1_b  = (const float*)d_in[2];
  const float* q_w    = (const float*)d_in[3];
  const float* q_b    = (const float*)d_in[4];
  const float* k_w    = (const float*)d_in[5];
  const float* k_b    = (const float*)d_in[6];
  const float* v_w    = (const float*)d_in[7];
  const float* v_b    = (const float*)d_in[8];
  const float* rpb    = (const float*)d_in[9];
  const int*   rpi    = (const int*)  d_in[10];
  const float* proj_w = (const float*)d_in[11];
  const float* proj_b = (const float*)d_in[12];
  const float* ln2_g  = (const float*)d_in[13];
  const float* ln2_b  = (const float*)d_in[14];
  const float* fc1_w  = (const float*)d_in[15];
  const float* fc1_b  = (const float*)d_in[16];
  const float* fc2_w  = (const float*)d_in[17];
  const float* fc2_b  = (const float*)d_in[18];
  float* out = (float*)d_out;

  // ---- workspace layout (bytes, 256-aligned sizes), with overlays ---------
  char* ws = (char*)d_ws;
  const size_t W33   = (size_t)DIM * DIM * 2;         // 294912
  const size_t WFFN  = (size_t)DIM * FFN * 2;         // 1179648
  const size_t XWB   = (size_t)MWROWS * DIM * 2;      // 25165824 (bf16)
  size_t o = 0;
  unsigned short* wtq  = (unsigned short*)(ws + o); o += W33;
  unsigned short* wtk  = (unsigned short*)(ws + o); o += W33;
  unsigned short* wtv  = (unsigned short*)(ws + o); o += W33;
  unsigned short* wtp  = (unsigned short*)(ws + o); o += W33;
  unsigned short* wtf1 = (unsigned short*)(ws + o); o += WFFN;
  unsigned short* wtf2 = (unsigned short*)(ws + o); o += WFFN;
  unsigned short* xw   = (unsigned short*)(ws + o); o += XWB;
  unsigned short* qb   = (unsigned short*)(ws + o); o += XWB;
  unsigned short* kb   = (unsigned short*)(ws + o); o += XWB;
  unsigned short* vb   = (unsigned short*)(ws + o); o += XWB;
  unsigned short* ctx  = (unsigned short*)(ws + o); o += XWB;
  float*          hbuf = (float*)(ws + o);          o += (size_t)MROWS * DIM * 4;
  // overlays (lifetimes disjoint):
  float*          projf = (float*)qb;        // 50.3 MB == q+k region, after attn
  unsigned short* y2    = xw;                // LN2 out, after qkv consumed xw
  unsigned short* ffn1  = qb;                // 77 MB spans q+k+v+part of ctx

  // ---- 1. convert+transpose weights to bf16 [N][K] ------------------------
  {
    int n33 = DIM * DIM, nff = DIM * FFN;
    cvt_transpose_bf16<<<(n33 + 255) / 256, 256, 0, stream>>>(q_w,    wtq,  DIM, DIM);
    cvt_transpose_bf16<<<(n33 + 255) / 256, 256, 0, stream>>>(k_w,    wtk,  DIM, DIM);
    cvt_transpose_bf16<<<(n33 + 255) / 256, 256, 0, stream>>>(v_w,    wtv,  DIM, DIM);
    cvt_transpose_bf16<<<(n33 + 255) / 256, 256, 0, stream>>>(proj_w, wtp,  DIM, DIM);
    cvt_transpose_bf16<<<(nff + 255) / 256, 256, 0, stream>>>(fc1_w,  wtf1, DIM, FFN);
    cvt_transpose_bf16<<<(nff + 255) / 256, 256, 0, stream>>>(fc2_w,  wtf2, FFN, DIM);
  }

  // ---- 2. LN1 + roll + window partition -----------------------------------
  ln1_window_kernel<<<NWIN * NPAD, 128, 0, stream>>>(hs, ln1_g, ln1_b, xw);

  // ---- 3. QKV projections: (32768/64)*(384/64)=3072 macro-tiles /8 waves --
  {
    int blocks = (MWROWS / 64) * (DIM / 64) / 8;   // 384
    gemm_bf16_wmma<<<blocks, 256, 0, stream>>>(xw, wtq, q_b, qb, nullptr,
                                               MWROWS, DIM, DIM, GEMM_OUT_BF16);
    gemm_bf16_wmma<<<blocks, 256, 0, stream>>>(xw, wtk, k_b, kb, nullptr,
                                               MWROWS, DIM, DIM, GEMM_OUT_BF16);
    gemm_bf16_wmma<<<blocks, 256, 0, stream>>>(xw, wtv, v_b, vb, nullptr,
                                               MWROWS, DIM, DIM, GEMM_OUT_BF16);
  }

  // ---- 4. windowed attention (512 windows x 12 heads) ---------------------
  attn_kernel<<<dim3(NWIN, NH), 128, 0, stream>>>(qb, kb, vb, rpb, rpi, ctx);

  // ---- 5. output projection -> f32 (scattered later) ----------------------
  {
    int blocks = (MWROWS / 64) * (DIM / 64) / 8;   // 384
    gemm_bf16_wmma<<<blocks, 256, 0, stream>>>(ctx, wtp, proj_b, projf, nullptr,
                                               MWROWS, DIM, DIM, GEMM_OUT_F32);
  }

  // ---- 6. window-reverse + roll + residual -> h; LN2 -> y2 ----------------
  gather_ln2_kernel<<<MROWS, 128, 0, stream>>>(hs, projf, ln2_g, ln2_b, hbuf, y2);

  // ---- 7. FC1 + exact GELU (M=25088, N=1536, K=384) -----------------------
  {
    int blocks = (MROWS / 64) * (FFN / 64) / 8;    // 392*24/8 = 1176
    gemm_bf16_wmma<<<blocks, 256, 0, stream>>>(y2, wtf1, fc1_b, ffn1, nullptr,
                                               MROWS, FFN, DIM, GEMM_OUT_BF16_GELU);
  }

  // ---- 8. FC2 + residual -> d_out (M=25088, N=384, K=1536) ----------------
  {
    int blocks = (MROWS / 64) * (DIM / 64) / 8;    // 392*6/8 = 294
    gemm_bf16_wmma<<<blocks, 256, 0, stream>>>(ffn1, wtf2, fc2_b, out, hbuf,
                                               MROWS, DIM, FFN, GEMM_OUT_F32_RES);
  }
}
